// FCT_53850299958032
// MI455X (gfx1250) — compile-verified, matrix operand order checked by reading, not run on hardware
//
#include <hip/hip_runtime.h>
#include <cmath>

// ---------------------------------------------------------------------------
// MI455X (gfx1250): full pipeline (rfft2/irfft2 as DFT-basis matmuls) on one
// templated bf16 WMMA GEMM (v_wmma_f32_16x16x32_bf16, wave32, 128x128 tile,
// K-step 32, double-buffered LDS). All GEMM operands live in global memory as
// bf16 (converted once / generated directly), so the staging path is pure
// b128 load -> b128 LDS store with zero conversion VALU in the hot loop.
// Workspace: ~715 MB, static byte offsets, sequenced aliasing.
// ---------------------------------------------------------------------------

#define TPB 256

typedef __attribute__((ext_vector_type(16))) __bf16 v16bf;
typedef __attribute__((ext_vector_type(8)))  __bf16 v8bf;
typedef __attribute__((ext_vector_type(8)))  float  v8f;

__device__ __forceinline__ unsigned short bfb(float f) {
  unsigned u = __builtin_bit_cast(unsigned, f);
  u += 0x7FFFu + ((u >> 16) & 1u);            // round-to-nearest-even
  return (unsigned short)(u >> 16);
}
__device__ __forceinline__ __bf16 f2bf(float f) {
  unsigned short h = bfb(f);
  return __builtin_bit_cast(__bf16, h);
}

// ---------------------------------------------------------------------------
// C[b] = scale * op(A[b]) @ op(B[b]) (*E[b]) (+bias) + beta*C[b]
// A,B bf16 (lda/ldb multiples of 8, 16B-aligned bases). C written to bf16
// (Cb) and/or f32 (Cf); beta-accumulation reads whichever exists.
// ---------------------------------------------------------------------------
template<bool TA, bool TB>
__global__ __launch_bounds__(TPB) void gemm_bf16(
    int M, int N, int K,
    const __bf16* __restrict__ A, int lda, long long sA,
    const __bf16* __restrict__ B, int ldb, long long sB,
    __bf16* __restrict__ Cb, float* __restrict__ Cf, int ldc, long long sC,
    const float* __restrict__ E, int ldE, long long sE,
    const float* __restrict__ bias,
    float scale, float beta)
{
  __shared__ alignas(16) __bf16 As[2][128 * 40];   // [m][k], 80B rows
  __shared__ alignas(16) __bf16 Bs[2][128 * 40];   // [n][k]

  const int b = blockIdx.z;
  A += (long long)b * sA;
  B += (long long)b * sB;
  if (Cb) Cb += (long long)b * sC;
  if (Cf) Cf += (long long)b * sC;
  if (E)  E  += (long long)b * sE;

  const int tid  = threadIdx.x;
  const int m0   = blockIdx.y * 128;
  const int n0   = blockIdx.x * 128;
  const int lane = tid & 31;
  const int w    = tid >> 5;
  const int wy   = (w & 1) * 64;
  const int wx   = (w >> 1) * 32;
  const int lr   = lane & 15;
  const int kh   = (lane >> 4) * 8;
  const __bf16 ZB = __builtin_bit_cast(__bf16, (unsigned short)0);

  v8f acc[4][2];
#pragma unroll
  for (int i = 0; i < 4; ++i)
#pragma unroll
    for (int j = 0; j < 2; ++j)
      acc[i][j] = (v8f){0.f, 0.f, 0.f, 0.f, 0.f, 0.f, 0.f, 0.f};

  const int nkt = (K + 31) >> 5;

  auto load_tile = [&](int t, int buf) {
    const int k0 = t << 5;
    if (k0 + 32 <= K) {
      // fast path: 512 chunks of 8 bf16 (16B), pure moves
#pragma unroll
      for (int i = 0; i < 2; ++i) {
        const int chunk = tid + i * TPB;            // 0..511
        // ---- A ----
        if (!TA) {                                   // A[m][k], k contiguous
          int ml = chunk >> 2, kc = (chunk & 3) << 3;
          int gm = m0 + ml;  if (gm >= M) gm = M - 1;   // bad rows masked later
          *(v8bf*)&As[buf][ml * 40 + kc] =
              *(const v8bf*)(A + (size_t)gm * lda + (k0 + kc));
          if (i == 0 && k0 + 96 <= K)
            __builtin_prefetch((const void*)(A + (size_t)gm * lda + (k0 + 64 + kc)), 0, 1);
        } else {                                     // A[k][m], m contiguous
          int kl = chunk >> 4, mc = (chunk & 15) << 3;
          int gm = m0 + mc;
          const __bf16* rowp = A + (size_t)(k0 + kl) * lda;
          __bf16* p = &As[buf][mc * 40 + kl];
          if (gm + 8 <= M) {
            const v8bf v = *(const v8bf*)(rowp + gm);
#pragma unroll
            for (int e = 0; e < 8; ++e) p[e * 40] = v[e];
          } else {
#pragma unroll
            for (int e = 0; e < 8; ++e) {
              int m = gm + e; int cm = m < M ? m : M - 1;
              __bf16 t2 = rowp[cm];
              p[e * 40] = (m < M) ? t2 : ZB;
            }
          }
        }
        // ---- B ----
        if (TB) {                                    // B[n][k], k contiguous
          int nl = chunk >> 2, kc = (chunk & 3) << 3;
          int gn = n0 + nl;  if (gn >= N) gn = N - 1;
          *(v8bf*)&Bs[buf][nl * 40 + kc] =
              *(const v8bf*)(B + (size_t)gn * ldb + (k0 + kc));
          if (i == 0 && k0 + 96 <= K)
            __builtin_prefetch((const void*)(B + (size_t)gn * ldb + (k0 + 64 + kc)), 0, 1);
        } else {                                     // B[k][n], n contiguous
          int kl = chunk >> 4, nc = (chunk & 15) << 3;
          int gn = n0 + nc;
          const __bf16* rowp = B + (size_t)(k0 + kl) * ldb;
          __bf16* p = &Bs[buf][nc * 40 + kl];
          if (gn + 8 <= N) {
            const v8bf v = *(const v8bf*)(rowp + gn);
#pragma unroll
            for (int e = 0; e < 8; ++e) p[e * 40] = v[e];
          } else {
#pragma unroll
            for (int e = 0; e < 8; ++e) {
              int n = gn + e; int cn = n < N ? n : N - 1;
              __bf16 t2 = rowp[cn];
              p[e * 40] = (n < N) ? t2 : ZB;
            }
          }
        }
      }
    } else {
      // K-edge (last tile only): scalar clamped loads + selects, no exec branches
#pragma unroll
      for (int i = 0; i < 16; ++i) {
        int idx = tid + i * TPB;
        int ml, kl;
        if (!TA) { ml = idx >> 5;  kl = idx & 31; }
        else     { ml = idx & 127; kl = idx >> 7; }
        int gm = m0 + ml, gk = k0 + kl;
        int cm = gm < M ? gm : M - 1;
        int ck = gk < K ? gk : K - 1;
        __bf16 va = TA ? A[(size_t)ck * lda + cm] : A[(size_t)cm * lda + ck];
        As[buf][ml * 40 + kl] = (gk < K) ? va : ZB;

        int nl, kl2;
        if (!TB) { nl = idx & 127; kl2 = idx >> 7; }
        else     { nl = idx >> 5;  kl2 = idx & 31; }
        int gn = n0 + nl, gk2 = k0 + kl2;
        int cn = gn < N ? gn : N - 1;
        int ck2 = gk2 < K ? gk2 : K - 1;
        __bf16 vb = TB ? B[(size_t)cn * ldb + ck2] : B[(size_t)ck2 * ldb + cn];
        Bs[buf][nl * 40 + kl2] = (gk2 < K) ? vb : ZB;
      }
    }
  };

  auto compute = [&](int buf) {
    v16bf af[4], bf[2];
#pragma unroll
    for (int i = 0; i < 4; ++i) {
      const __bf16* p = &As[buf][(wy + i * 16 + lr) * 40 + kh];
      ((v8bf*)&af[i])[0] = *(const v8bf*)p;         // k = kh+0..7
      ((v8bf*)&af[i])[1] = *(const v8bf*)(p + 16);  // k = kh+16..23
    }
#pragma unroll
    for (int j = 0; j < 2; ++j) {
      const __bf16* p = &Bs[buf][(wx + j * 16 + lr) * 40 + kh];
      ((v8bf*)&bf[j])[0] = *(const v8bf*)p;
      ((v8bf*)&bf[j])[1] = *(const v8bf*)(p + 16);
    }
#pragma unroll
    for (int i = 0; i < 4; ++i)
#pragma unroll
      for (int j = 0; j < 2; ++j)
        acc[i][j] = __builtin_amdgcn_wmma_f32_16x16x32_bf16(
            false, af[i], false, bf[j], (short)0, acc[i][j], false, false);
  };

  load_tile(0, 0);
  __syncthreads();
  for (int t = 0; t < nkt; ++t) {
    const int cur = t & 1;
    if (t + 1 < nkt) load_tile(t + 1, cur ^ 1);
    compute(cur);
    __syncthreads();
  }

  // Epilogue. C layout: VGPR r -> M = r + (lane>=16 ? 8 : 0); N = lane%16.
#pragma unroll
  for (int i = 0; i < 4; ++i) {
#pragma unroll
    for (int j = 0; j < 2; ++j) {
      int n = n0 + wx + j * 16 + lr;
      bool nok = n < N;
#pragma unroll
      for (int r = 0; r < 8; ++r) {
        int m = m0 + wy + i * 16 + r + kh;
        if (nok && m < M) {
          size_t off = (size_t)m * ldc + n;
          float v = scale * acc[i][j][r];
          if (E)           v *= E[(size_t)m * ldE + n];
          if (bias)        v += bias[n];
          if (beta != 0.f) v += beta * (Cf ? Cf[off] : (float)Cb[off]);
          if (Cf) Cf[off] = v;
          if (Cb) Cb[off] = __builtin_bit_cast(__bf16, bfb(v));
        }
      }
    }
  }
}

// ---------------------------------------------------------------------------
// f32 -> bf16 bulk conversion (float4 in, 8B out); n must be a multiple of 4
// ---------------------------------------------------------------------------
__global__ void cvt_bf16_v4(const float* __restrict__ s, __bf16* __restrict__ d,
                            long long n4) {
  long long i = (long long)blockIdx.x * blockDim.x + threadIdx.x;
  if (i >= n4) return;
  const float4 v = ((const float4*)s)[i];
  uint2 r;
  r.x = (unsigned)bfb(v.x) | ((unsigned)bfb(v.y) << 16);
  r.y = (unsigned)bfb(v.z) | ((unsigned)bfb(v.w) << 16);
  ((uint2*)d)[i] = r;
}

// ---------------------------------------------------------------------------
// DFT basis generation, directly in bf16 (exact integer angle reduction)
// ---------------------------------------------------------------------------
__global__ void gen_dft_chan(__bf16* __restrict__ CCm, __bf16* __restrict__ SCm,
                             __bf16* __restrict__ CWm, __bf16* __restrict__ SWm) {
  int idx = blockIdx.x * blockDim.x + threadIdx.x;     // 1025*2048
  if (idx >= 1025 * 2048) return;
  int v = idx >> 11, c = idx & 2047;
  int t = (v * c) & 2047;
  float ang = (float)t * (6.283185307179586f / 2048.0f);
  float s, co;
  sincosf(ang, &s, &co);
  float wv = (v == 0 || v == 1024) ? 1.f : 2.f;        // irfft hermitian weights
  CCm[idx] = f2bf(co);       SCm[idx] = f2bf(s);
  CWm[idx] = f2bf(wv * co);  SWm[idx] = f2bf(wv * s);
}

__global__ void gen_dft_spat(__bf16* __restrict__ CMm, __bf16* __restrict__ SMm) {
  int idx = blockIdx.x * blockDim.x + threadIdx.x;     // 400*400
  if (idx >= 160000) return;
  int u = idx / 400, m = idx % 400;
  int t = (u * m) % 400;
  float ang = (float)t * (6.283185307179586f / 400.0f);
  float s, co;
  sincosf(ang, &s, &co);
  CMm[idx] = f2bf(co);  SMm[idx] = f2bf(s);
}

// ---------------------------------------------------------------------------
// logmax: L = log(a)/rowsum(log(a)); reads/scratches f32 a, emits bf16 L
// grid = (rows, B)
// ---------------------------------------------------------------------------
__global__ void logmax_rows(float* __restrict__ a, __bf16* __restrict__ ob,
                            long long bStride, int ld, int cols) {
  size_t base = (size_t)blockIdx.y * bStride + (size_t)blockIdx.x * ld;
  float* row = a + base;
  __bf16* orow = ob + base;
  __shared__ float red[TPB];
  float s = 0.f;
  for (int i = threadIdx.x; i < cols; i += TPB) {
    float l = __logf(row[i]);
    row[i] = l;
    s += l;
  }
  red[threadIdx.x] = s;
  __syncthreads();
  for (int o = TPB / 2; o > 0; o >>= 1) {
    if (threadIdx.x < o) red[threadIdx.x] += red[threadIdx.x + o];
    __syncthreads();
  }
  float inv = 1.0f / red[0];
  for (int i = threadIdx.x; i < cols; i += TPB)
    orow[i] = f2bf(row[i] * inv);
}

// ---------------------------------------------------------------------------
// BatchNorm over relu(z) with the reference's raw reshape (channel = flat/400)
// ---------------------------------------------------------------------------
__global__ void bn_stats(const float* __restrict__ z, float* __restrict__ stats) {
  const int c = blockIdx.x;
  __shared__ float r1[TPB], r2[TPB];
  float s = 0.f, ss = 0.f;
  for (int i = threadIdx.x; i < 32 * 400; i += TPB) {
    int b = i / 400, p = i % 400;
    float v = z[(size_t)b * 819200 + (size_t)c * 400 + p];
    v = fmaxf(v, 0.f);
    s += v; ss += v * v;
  }
  r1[threadIdx.x] = s;
  r2[threadIdx.x] = ss;
  __syncthreads();
  for (int o = TPB / 2; o > 0; o >>= 1) {
    if (threadIdx.x < o) { r1[threadIdx.x] += r1[threadIdx.x + o]; r2[threadIdx.x] += r2[threadIdx.x + o]; }
    __syncthreads();
  }
  if (threadIdx.x == 0) {
    float mu  = r1[0] * (1.f / 12800.f);
    float var = r2[0] * (1.f / 12800.f) - mu * mu;
    stats[c] = mu;
    stats[2048 + c] = rsqrtf(var + 1e-5f);
  }
}

__global__ void bn_apply(const float* __restrict__ x, const float* __restrict__ z,
                         const float* __restrict__ stats,
                         const float* __restrict__ gamma, const float* __restrict__ beta,
                         float* __restrict__ out) {
  size_t idx = (size_t)blockIdx.x * blockDim.x + threadIdx.x;
  if (idx >= (size_t)32 * 819200) return;
  int r = (int)(idx % 819200);
  int c = r / 400;
  float v = fmaxf(z[idx], 0.f);
  out[idx] = x[idx] + gamma[c] * (v - stats[c]) * stats[2048 + c] + beta[c];
}

// ---------------------------------------------------------------------------
// Host orchestration
// ---------------------------------------------------------------------------
static constexpr int       BB    = 32;
static constexpr int       CH    = 2048;
static constexpr int       NSP   = 400;
static constexpr int       FF    = 1025;
static constexpr int       FFP   = 1032;                  // padded (mult of 8)
static constexpr long long S_BC  = (long long)NSP * CH;   // 819200 elems
static constexpr long long S_BFP = (long long)NSP * FFP;  // 412800 elems
static constexpr long long S_X   = (long long)FF * NSP;   // 410000 elems
static constexpr long long S_FAP = (long long)FF * FFP;   // 1057800 elems

// byte offsets into workspace (all multiples of 16)
static constexpr long long SZ_FCb = (long long)FF * CH * 2;     // 4198400
static constexpr long long oCC  = 0;
static constexpr long long oSC  = oCC + SZ_FCb;
static constexpr long long oCW  = oSC + SZ_FCb;
static constexpr long long oSW  = oCW + SZ_FCb;
static constexpr long long oCM  = oSW + SZ_FCb;
static constexpr long long oSM  = oCM + 320000;
static constexpr long long oXB  = oSM + 320000;                  // x bf16
static constexpr long long oWQ  = oXB + BB * S_BC * 2;
static constexpr long long oWK  = oWQ + 320000;
static constexpr long long oWP  = oWK + 320000;
static constexpr long long oQ   = oWP + (long long)CH * CH * 2;
static constexpr long long oK   = oQ  + BB * S_BC * 2;
static constexpr long long oSR  = oK  + BB * S_BC * 2;
static constexpr long long oSI  = oSR + BB * S_BFP * 2;
static constexpr long long oQFR = oSI + BB * S_BFP * 2;
static constexpr long long oQFI = oQFR + BB * S_BFP * 2;
static constexpr long long oKFR = oQFI + BB * S_BFP * 2;
static constexpr long long oKFI = oKFR + BB * S_BFP * 2;
static constexpr long long oAR  = oKFI + BB * S_BFP * 2;         // f32
static constexpr long long oAI  = oAR + BB * S_FAP * 4;          // f32
static constexpr long long oLR  = oAI + BB * S_FAP * 4;          // bf16
static constexpr long long oLI  = oLR + BB * S_FAP * 2;
static constexpr long long oST  = oLI + BB * S_FAP * 2;          // f32 stats
// aliases (lifetimes strictly sequenced on the stream):
static constexpr long long oY  = oQ;                 // q dead after its spectrum
static constexpr long long oXR = oK;                 // k dead after its spectrum
static constexpr long long oXI = oK + BB * S_X * 2;  // spills into dead SR region
static constexpr long long oTR = oAI;                // AI dead after logmax
static constexpr long long oTI = oAI + BB * S_BFP * 2;
static constexpr long long oZ  = oAR;                // AR dead after logmax (f32)

template<bool TA, bool TB>
static inline void G(hipStream_t st, int M, int N, int K,
                     const __bf16* A, int lda, long long sA,
                     const __bf16* B, int ldb, long long sB,
                     __bf16* Cb, float* Cf, int ldc, long long sC,
                     const float* E, int ldE, long long sE,
                     const float* bias, float scale, float beta) {
  dim3 g((N + 127) / 128, (M + 127) / 128, BB), blk(TPB);
  gemm_bf16<TA, TB><<<g, blk, 0, st>>>(M, N, K, A, lda, sA, B, ldb, sB,
                                       Cb, Cf, ldc, sC, E, ldE, sE, bias, scale, beta);
}

extern "C" void kernel_launch(void* const* d_in, const int* in_sizes, int n_in,
                              void* d_out, int out_size, void* d_ws, size_t ws_size,
                              hipStream_t stream) {
  const float* x     = (const float*)d_in[0];
  const float* Wq    = (const float*)d_in[1];
  const float* Wk    = (const float*)d_in[2];
  const float* alpha = (const float*)d_in[3];
  const float* Wp    = (const float*)d_in[4];
  const float* bp    = (const float*)d_in[5];
  const float* gamma = (const float*)d_in[6];
  const float* beta  = (const float*)d_in[7];
  float* out = (float*)d_out;

  char* ws = (char*)d_ws;
  __bf16 *CCb = (__bf16*)(ws + oCC), *SCb = (__bf16*)(ws + oSC);
  __bf16 *CWb = (__bf16*)(ws + oCW), *SWb = (__bf16*)(ws + oSW);
  __bf16 *CMb = (__bf16*)(ws + oCM), *SMb = (__bf16*)(ws + oSM);
  __bf16 *xb  = (__bf16*)(ws + oXB);
  __bf16 *Wqb = (__bf16*)(ws + oWQ), *Wkb = (__bf16*)(ws + oWK), *Wpb = (__bf16*)(ws + oWP);
  __bf16 *Qb  = (__bf16*)(ws + oQ),  *Kbb = (__bf16*)(ws + oK);
  __bf16 *SRb = (__bf16*)(ws + oSR), *SIb = (__bf16*)(ws + oSI);
  __bf16 *QFRb = (__bf16*)(ws + oQFR), *QFIb = (__bf16*)(ws + oQFI);
  __bf16 *KFRb = (__bf16*)(ws + oKFR), *KFIb = (__bf16*)(ws + oKFI);
  float  *AR = (float*)(ws + oAR), *AI = (float*)(ws + oAI);
  __bf16 *Lr = (__bf16*)(ws + oLR), *Li = (__bf16*)(ws + oLI);
  __bf16 *XRb = (__bf16*)(ws + oXR), *XIb = (__bf16*)(ws + oXI);
  __bf16 *TRb = (__bf16*)(ws + oTR), *TIb = (__bf16*)(ws + oTI);
  __bf16 *Yb  = (__bf16*)(ws + oY);
  float  *Z = (float*)(ws + oZ), *ST = (float*)(ws + oST);

  const float SF = 1.0f / sqrtf((float)(NSP * CH));   // ortho 1/sqrt(819200)

  // bases + one-time input conversions
  gen_dft_chan<<<(int)((FF * CH) / TPB), TPB, 0, stream>>>(CCb, SCb, CWb, SWb);
  gen_dft_spat<<<160000 / TPB, TPB, 0, stream>>>(CMb, SMb);
  cvt_bf16_v4<<<(int)((BB * S_BC / 4 + TPB - 1) / TPB), TPB, 0, stream>>>(x,  xb,  BB * S_BC / 4);
  cvt_bf16_v4<<<(int)((160000 / 4 + TPB - 1) / TPB), TPB, 0, stream>>>(Wq, Wqb, 160000 / 4);
  cvt_bf16_v4<<<(int)((160000 / 4 + TPB - 1) / TPB), TPB, 0, stream>>>(Wk, Wkb, 160000 / 4);
  cvt_bf16_v4<<<(int)(((long long)CH * CH / 4 + TPB - 1) / TPB), TPB, 0, stream>>>(Wp, Wpb, (long long)CH * CH / 4);

  // q = Wq @ x^T, k = Wk @ x^T          [400,2048] per batch (NT)
  G<false, true>(stream, NSP, CH, NSP, Wqb, NSP, 0, xb, NSP, S_BC, Qb,  nullptr, CH, S_BC, nullptr, 0, 0, nullptr, 1.f, 0.f);
  G<false, true>(stream, NSP, CH, NSP, Wkb, NSP, 0, xb, NSP, S_BC, Kbb, nullptr, CH, S_BC, nullptr, 0, 0, nullptr, 1.f, 0.f);

  // rfft2 as matmuls: S = q @ basis^T (channel axis), then spatial DFT
  auto spectrum = [&](const __bf16* src, __bf16* FRo, __bf16* FIo) {
    G<false, true >(stream, NSP, FF, CH,  src, CH, S_BC, CCb, CH, 0,     SRb, nullptr, FFP, S_BFP, nullptr, 0, 0, nullptr,  1.f, 0.f);
    G<false, true >(stream, NSP, FF, CH,  src, CH, S_BC, SCb, CH, 0,     SIb, nullptr, FFP, S_BFP, nullptr, 0, 0, nullptr, -1.f, 0.f);
    G<false, false>(stream, NSP, FF, NSP, CMb, NSP, 0,   SRb, FFP, S_BFP, FRo, nullptr, FFP, S_BFP, nullptr, 0, 0, nullptr,  SF, 0.f);
    G<false, false>(stream, NSP, FF, NSP, SMb, NSP, 0,   SIb, FFP, S_BFP, FRo, nullptr, FFP, S_BFP, nullptr, 0, 0, nullptr,  SF, 1.f);
    G<false, false>(stream, NSP, FF, NSP, CMb, NSP, 0,   SIb, FFP, S_BFP, FIo, nullptr, FFP, S_BFP, nullptr, 0, 0, nullptr,  SF, 0.f);
    G<false, false>(stream, NSP, FF, NSP, SMb, NSP, 0,   SRb, FFP, S_BFP, FIo, nullptr, FFP, S_BFP, nullptr, 0, 0, nullptr, -SF, 1.f);
  };
  spectrum(Qb,  QFRb, QFIb);
  spectrum(Kbb, KFRb, KFIb);

  // attn = alpha .* (qf^T @ kf)         [1025,1025] per batch (TN, fused alpha, f32 out)
  G<true, false>(stream, FF, FF, NSP, QFRb, FFP, S_BFP, KFRb, FFP, S_BFP, nullptr, AR, FFP, S_FAP, alpha, FF, 0, nullptr, 1.f, 0.f);
  G<true, false>(stream, FF, FF, NSP, QFIb, FFP, S_BFP, KFIb, FFP, S_BFP, nullptr, AI, FFP, S_FAP, alpha, FF, 0, nullptr, 1.f, 0.f);

  // logmax -> bf16 L
  logmax_rows<<<dim3(FF, BB), TPB, 0, stream>>>(AR, Lr, S_FAP, FFP, FF);
  logmax_rows<<<dim3(FF, BB), TPB, 0, stream>>>(AI, Li, S_FAP, FFP, FF);

  // X = L @ qf^T                        [1025,400] per batch (NT)
  G<false, true>(stream, FF, NSP, FF, Lr, FFP, S_FAP, QFIb, FFP, S_BFP, XRb, nullptr, NSP, S_X, nullptr, 0, 0, nullptr, 1.f, 0.f);
  G<false, true>(stream, FF, NSP, FF, Li, FFP, S_FAP, QFRb, FFP, S_BFP, XIb, nullptr, NSP, S_X, nullptr, 0, 0, nullptr, 1.f, 0.f);

  // inverse spatial DFT: Tr = Cm@Xr^T - Sm@Xi^T ; Ti = Sm@Xr^T + Cm@Xi^T
  G<false, true>(stream, NSP, FF, NSP, CMb, NSP, 0, XRb, NSP, S_X, TRb, nullptr, FFP, S_BFP, nullptr, 0, 0, nullptr,  1.f, 0.f);
  G<false, true>(stream, NSP, FF, NSP, SMb, NSP, 0, XIb, NSP, S_X, TRb, nullptr, FFP, S_BFP, nullptr, 0, 0, nullptr, -1.f, 1.f);
  G<false, true>(stream, NSP, FF, NSP, SMb, NSP, 0, XRb, NSP, S_X, TIb, nullptr, FFP, S_BFP, nullptr, 0, 0, nullptr,  1.f, 0.f);
  G<false, true>(stream, NSP, FF, NSP, CMb, NSP, 0, XIb, NSP, S_X, TIb, nullptr, FFP, S_BFP, nullptr, 0, 0, nullptr,  1.f, 1.f);

  // inverse channel DFT (irfft weights in CW/SW): y = SF*(Tr@CW - Ti@SW)
  G<false, false>(stream, NSP, CH, FF, TRb, FFP, S_BFP, CWb, CH, 0, Yb, nullptr, CH, S_BC, nullptr, 0, 0, nullptr,  SF, 0.f);
  G<false, false>(stream, NSP, CH, FF, TIb, FFP, S_BFP, SWb, CH, 0, Yb, nullptr, CH, S_BC, nullptr, 0, 0, nullptr, -SF, 1.f);

  // projection: z = y @ Wp^T + bp       (NT, fused bias, f32 out)
  G<false, true>(stream, NSP, CH, CH, Yb, CH, S_BC, Wpb, CH, 0, nullptr, Z, CH, S_BC, nullptr, 0, 0, bp, 1.f, 0.f);

  // relu + batchnorm (training stats) + residual
  bn_stats<<<CH, TPB, 0, stream>>>(Z, ST);
  bn_apply<<<(int)((BB * S_BC + TPB - 1) / TPB), TPB, 0, stream>>>(x, Z, ST, gamma, beta, out);
  (void)in_sizes; (void)n_in; (void)out_size; (void)ws_size;
}
